// PNet_33002528702638
// MI455X (gfx1250) — matrix operand order repaired and checked by reading, not compiled, not run on hardware
//
#include <hip/hip_runtime.h>

typedef __attribute__((ext_vector_type(16))) _Float16 v16h;
typedef __attribute__((ext_vector_type(8)))  _Float16 v8h;
typedef __attribute__((ext_vector_type(4)))  _Float16 v4h;
typedef __attribute__((ext_vector_type(8)))  float    v8f;

#define NUM_NODES 65536
#define NGRAPH    1024
#define NPG       64        // nodes per graph
#define HID       256
#define H1        128
#define NEDGE     2097152

static __device__ __forceinline__ v16h concat8(v8h a, v8h b) {
    return __builtin_shufflevector(a, b, 0,1,2,3,4,5,6,7,8,9,10,11,12,13,14,15);
}

// ---------------------------------------------------------------- kernels ---

__global__ void zero_deg_kernel(unsigned* __restrict__ deg) {
    int i = blockIdx.x * blockDim.x + threadIdx.x;
    if (i < NUM_NODES) deg[i] = 0u;
}

__global__ void hist_kernel(const int* __restrict__ src, unsigned* __restrict__ deg) {
    int stride = gridDim.x * blockDim.x;
    for (int e = blockIdx.x * blockDim.x + threadIdx.x; e < NEDGE; e += stride)
        atomicAdd(&deg[src[e]], 1u);
}

// W1t[n][k] = W1[k+1][n]  (f16, 128x256) ; W2t[n][k] = W2[k][n] (f16, 128x128)
__global__ void prepw_kernel(const float* __restrict__ W1, const float* __restrict__ W2,
                             _Float16* __restrict__ W1t, _Float16* __restrict__ W2t) {
    int idx = blockIdx.x * blockDim.x + threadIdx.x;
    if (idx < H1 * HID) {
        int n = idx >> 8, k = idx & 255;
        W1t[idx] = (_Float16)W1[(k + 1) * H1 + n];
    } else {
        int i = idx - H1 * HID;
        if (i < H1 * H1) {
            int n = i >> 7, k = i & 127;
            W2t[i] = (_Float16)W2[k * H1 + n];
        }
    }
}

// stable per-graph rank by degree (lexsort semantics: ties -> original order)
__global__ void conn_kernel(const unsigned* __restrict__ deg, float* __restrict__ conn) {
    __shared__ unsigned dS[NPG];
    int g = blockIdx.x, j = threadIdx.x;
    unsigned dj = deg[g * NPG + j];
    dS[j] = dj;
    __syncthreads();
    int rank = 0;
    for (int k = 0; k < NPG; ++k) {
        unsigned dk = dS[k];
        rank += (dk < dj) || (dk == dj && k < j);
    }
    // conn[i] = (original local index of rank-(i%64) node) / 64
    conn[g * NPG + rank] = (float)j * (1.0f / 64.0f);
}

__global__ __launch_bounds__(256)
void mlp_softmax_kernel(const float* __restrict__ nf,
                        const float* __restrict__ W1,   // for row 0 (conn column)
                        const float* __restrict__ b1,
                        const float* __restrict__ b2,
                        const float* __restrict__ W3,
                        const float* __restrict__ b3,
                        const float* __restrict__ kT,
                        const float* __restrict__ conn,
                        const _Float16* __restrict__ W1t,
                        const _Float16* __restrict__ W2t,
                        float* __restrict__ out) {
    // padded strides: 264*2B=528 (33*16) and 136*2B=272 (17*16) -> bank rotation
    __shared__ __align__(16) _Float16 Axt[NPG * 264];   // features f16, 64x256
    __shared__ __align__(16) _Float16 Hs [NPG * 136];   // silu(h1) f16, 64x128
    __shared__ __align__(16) _Float16 H2s[NPG * 136];   // silu(h2) f16, 64x128
    __shared__ float connS[NPG];
    __shared__ float logitS[NPG];

    const int g    = blockIdx.x;
    const int tid  = threadIdx.x;
    const int lane = tid & 31;
    const int wave = tid >> 5;
    const int lnlo = lane & 15;
    const int hiK   = (lane >> 4) ? 8  : 0;   // A-fragment K offset (upper half-wave)
    const int hiK16 = (lane >> 4) ? 16 : 0;   // unused; B fragment also uses hiK
    const int hiM8  = (lane >> 4) ? 8  : 0;   // C-fragment M offset

    __builtin_prefetch(W1t, 0, 3);
    __builtin_prefetch(W2t, 0, 3);

    // ---- stage features (f32 global -> f16 LDS), coalesced float4 ----
    {
        const float4* nf4 = (const float4*)(nf + (size_t)g * NPG * HID);
        for (int i = tid; i < NPG * (HID / 4); i += 256) {
            int row = i >> 6, c4 = i & 63;
            float4 v = nf4[row * 64 + c4];
            v4h h;
            h[0] = (_Float16)v.x; h[1] = (_Float16)v.y;
            h[2] = (_Float16)v.z; h[3] = (_Float16)v.w;
            *(v4h*)&Axt[row * 264 + c4 * 4] = h;
        }
        if (tid < NPG) connS[tid] = conn[g * NPG + tid];
    }
    __syncthreads();

    const int n0   = wave * 16;       // 8 waves cover 128 output cols
    const int ncol = n0 + lnlo;
    const float bias1 = b1[ncol];
    const float bias2 = b2[ncol];
    const float w1r0  = W1[ncol];     // W1[0][ncol] : conn-column weight

    // ---- GEMM1: [64x256]f16 @ [256x128]f16 -> f32, + conn rank-1, SiLU ----
    for (int mt = 0; mt < 4; ++mt) {
        v8f acc = {};
        const int arow = mt * 16 + lnlo;
        for (int kk = 0; kk < 8; ++kk) {
            const int ka = kk * 32 + hiK;
            v8h a0 = *(const v8h*)&Axt[arow * 264 + ka];
            v8h a1 = *(const v8h*)&Axt[arow * 264 + ka + 16];
            v16h Af = concat8(a0, a1);
            const v8h* bp = (const v8h*)&W1t[ncol * HID + kk * 32 + hiK];
            v16h Bf = concat8(bp[0], bp[2]);
            acc = __builtin_amdgcn_wmma_f32_16x16x32_f16(false, Af, false, Bf,
                                                         (short)0, acc, false, false);
        }
        #pragma unroll
        for (int r = 0; r < 8; ++r) {
            int m = mt * 16 + r + hiM8;
            float v = acc[r] + connS[m] * w1r0 + bias1;
            v = v / (1.0f + __expf(-v));            // silu
            Hs[m * 136 + ncol] = (_Float16)v;
        }
    }
    __syncthreads();

    // ---- GEMM2: [64x128]f16 @ [128x128]f16 -> f32, SiLU ----
    for (int mt = 0; mt < 4; ++mt) {
        v8f acc = {};
        const int arow = mt * 16 + lnlo;
        for (int kk = 0; kk < 4; ++kk) {
            const int ka = kk * 32 + hiK;
            v8h a0 = *(const v8h*)&Hs[arow * 136 + ka];
            v8h a1 = *(const v8h*)&Hs[arow * 136 + ka + 16];
            v16h Af = concat8(a0, a1);
            const v8h* bp = (const v8h*)&W2t[ncol * H1 + kk * 32 + hiK];
            v16h Bf = concat8(bp[0], bp[2]);
            acc = __builtin_amdgcn_wmma_f32_16x16x32_f16(false, Af, false, Bf,
                                                         (short)0, acc, false, false);
        }
        #pragma unroll
        for (int r = 0; r < 8; ++r) {
            int m = mt * 16 + r + hiM8;
            float v = acc[r] + bias2;
            v = v / (1.0f + __expf(-v));
            H2s[m * 136 + ncol] = (_Float16)v;
        }
    }
    __syncthreads();

    // ---- GEMV vs W3 + per-graph softmax over 64 nodes ----
    if (tid < NPG) {
        float dot = 0.0f;
        #pragma unroll 8
        for (int k = 0; k < H1; ++k)
            dot += (float)H2s[tid * 136 + k] * W3[k];
        logitS[tid] = (dot + b3[0]) / kT[0];
    }
    __syncthreads();
    if (tid < NPG) {
        float mx = -3.0e38f;
        for (int k = 0; k < NPG; ++k) mx = fmaxf(mx, logitS[k]);
        float s = 0.0f;
        for (int k = 0; k < NPG; ++k) s += __expf(logitS[k] - mx);
        out[g * NPG + tid] = __expf(logitS[tid] - mx) / s;
    }
}

// ----------------------------------------------------------------- launch ---

extern "C" void kernel_launch(void* const* d_in, const int* in_sizes, int n_in,
                              void* d_out, int out_size, void* d_ws, size_t ws_size,
                              hipStream_t stream) {
    const float* nf  = (const float*)d_in[0];
    const float* W1  = (const float*)d_in[1];
    const float* b1  = (const float*)d_in[2];
    const float* W2  = (const float*)d_in[3];
    const float* b2  = (const float*)d_in[4];
    const float* W3  = (const float*)d_in[5];
    const float* b3  = (const float*)d_in[6];
    const float* kT  = (const float*)d_in[7];
    const int*   edge = (const int*)d_in[8];
    // d_in[9] (batch) is repeat(arange(1024),64): structure folded into indexing.

    char* ws = (char*)d_ws;
    unsigned*  deg  = (unsigned*)(ws);                  // 65536 * 4B
    float*     conn = (float*)   (ws + 262144);         // 65536 * 4B
    _Float16*  W1t  = (_Float16*)(ws + 524288);         // 32768 * 2B
    _Float16*  W2t  = (_Float16*)(ws + 589824);         // 16384 * 2B
    float*     out  = (float*)d_out;

    zero_deg_kernel<<<NUM_NODES / 256, 256, 0, stream>>>(deg);
    prepw_kernel<<<(H1 * HID + H1 * H1) / 256, 256, 0, stream>>>(W1, W2, W1t, W2t);
    hist_kernel<<<2048, 256, 0, stream>>>(edge, deg);
    conn_kernel<<<NGRAPH, NPG, 0, stream>>>(deg, conn);
    mlp_softmax_kernel<<<NGRAPH, 256, 0, stream>>>(nf, W1, b1, b2, W3, b3, kT,
                                                   conn, W1t, W2t, out);
}
